// LayerNormLSTM_17695265259793
// MI455X (gfx1250) — matrix operand order repaired and checked by reading, not compiled, not run on hardware
//
#include <hip/hip_runtime.h>
#include <cstdint>
#include <cstddef>

// ---------------------------------------------------------------------------
// LayerNorm-LSTM (2 layers), B=64 T=512 D=512 H=1024, for MI455X (gfx1250).
// Persistent wave32 WMMA kernel: bf16 matmuls (v_wmma_f32_16x16x32_bf16) with
// fp32 accumulation + fp32 recurrent state; device-scope atomic grid barrier
// between the 4 dependent phases of each timestep. Weights are transposed and
// converted to bf16 once per call (28 MB -> lives in the 192 MB L2).
// ---------------------------------------------------------------------------

typedef __attribute__((ext_vector_type(16))) __bf16        bf16x16;
typedef __attribute__((ext_vector_type(8)))  float         f32x8;
typedef __attribute__((ext_vector_type(4)))  unsigned int  u32x4;

#define T_STEPS 512
#define BATCH   64
#define DIN     512
#define HID     1024
#define G4H     4096
#define NBLK    64     // persistent grid: 64 blocks (block b owns batch row b in LN phases)
#define NTHR    256    // 8 wave32 per block

union BFrag { bf16x16 v; u32x4 u[2]; };

__device__ __forceinline__ f32x8 wmma_bf16(bf16x16 a, bf16x16 b, f32x8 c) {
  // D = A(16x32 bf16) x B(32x16 bf16) + C(16x16 f32)
  return __builtin_amdgcn_wmma_f32_16x16x32_bf16(false, a, false, b, (short)0, c,
                                                 false, false);
}

__device__ __forceinline__ float sigmoidf_(float x) {
  return 1.0f / (1.0f + __expf(-x));
}

// Device-scope split barrier across the 64 persistent blocks.
__device__ __forceinline__ void grid_barrier(unsigned* bar, unsigned& epoch) {
  epoch += NBLK;
  __threadfence();            // release: every thread flushes its own stores
  __syncthreads();
  if (threadIdx.x == 0) {
    __hip_atomic_fetch_add(bar, 1u, __ATOMIC_RELEASE, __HIP_MEMORY_SCOPE_AGENT);
    while (__hip_atomic_load(bar, __ATOMIC_ACQUIRE, __HIP_MEMORY_SCOPE_AGENT) < epoch)
      __builtin_amdgcn_s_sleep(2);
  }
  __syncthreads();
  __threadfence();            // acquire: invalidate stale near caches
}

// gates[64,4096] = A1[64,K1] @ W1^T + A2[64,K2] @ W2^T
// W1/W2 stored transposed+bf16: [4096, K] row-major (contiguous B-fragments).
// Each wave: one 16-col tile (nt), two 16-row tiles (mt0, mt0+1) sharing every
// B fragment (halves weight traffic).
template <int K1, int K2>
__device__ __forceinline__ void gemm_pair(const __bf16* __restrict__ A1, int lda1,
                                          const __bf16* __restrict__ A2, int lda2,
                                          const __bf16* __restrict__ W1,
                                          const __bf16* __restrict__ W2,
                                          float* __restrict__ gates) {
  const int lane = threadIdx.x & 31;
  const int wv   = blockIdx.x * (NTHR / 32) + (threadIdx.x >> 5);  // 0..511
  const int nt   = wv >> 1;                                        // 0..255
  const int mt0  = (wv & 1) * 2;                                   // 0 or 2
  // A 16x32 bf16 layout: lane L(0-15): M=L, K[0:8]+K[16:24]; lane L+16: K[8:16]+K[24:32]
  const int aM = lane & 15, aK = (lane >> 4) * 8;
  // B 32x16 bf16 layout: lanes 0-15: col N=lane, K[0:8]+K[16:24]; lanes 16-31: K[8:16]+K[24:32]
  const int bN = lane & 15, bK = (lane >> 4) * 8;

  f32x8 acc0 = {};
  f32x8 acc1 = {};

  {
    const __bf16* wp = W1 + (size_t)(nt * 16 + bN) * K1 + bK;
    const __bf16* a0 = A1 + (size_t)(mt0 * 16 + aM) * lda1;
    const __bf16* a1 = a0 + (size_t)16 * lda1;
#pragma unroll 4
    for (int kb = 0; kb < K1; kb += 32) {
      __builtin_prefetch(wp + kb + 256, 0, 3);   // WGP-scope prefetch of weight stream
      BFrag b, fa0, fa1;
      b.u[0]   = *(const u32x4*)(wp + kb);
      b.u[1]   = *(const u32x4*)(wp + kb + 16);
      fa0.u[0] = *(const u32x4*)(a0 + kb + aK);
      fa0.u[1] = *(const u32x4*)(a0 + kb + 16 + aK);
      fa1.u[0] = *(const u32x4*)(a1 + kb + aK);
      fa1.u[1] = *(const u32x4*)(a1 + kb + 16 + aK);
      acc0 = wmma_bf16(fa0.v, b.v, acc0);
      acc1 = wmma_bf16(fa1.v, b.v, acc1);
    }
  }
  {
    const __bf16* wp = W2 + (size_t)(nt * 16 + bN) * K2 + bK;
    const __bf16* a0 = A2 + (size_t)(mt0 * 16 + aM) * lda2;
    const __bf16* a1 = a0 + (size_t)16 * lda2;
#pragma unroll 4
    for (int kb = 0; kb < K2; kb += 32) {
      __builtin_prefetch(wp + kb + 256, 0, 3);
      BFrag b, fa0, fa1;
      b.u[0]   = *(const u32x4*)(wp + kb);
      b.u[1]   = *(const u32x4*)(wp + kb + 16);
      fa0.u[0] = *(const u32x4*)(a0 + kb + aK);
      fa0.u[1] = *(const u32x4*)(a0 + kb + 16 + aK);
      fa1.u[0] = *(const u32x4*)(a1 + kb + aK);
      fa1.u[1] = *(const u32x4*)(a1 + kb + 16 + aK);
      acc0 = wmma_bf16(fa0.v, b.v, acc0);
      acc1 = wmma_bf16(fa1.v, b.v, acc1);
    }
  }

  // C/D 16x16 f32 layout: VGPR r -> (M = r + 8*(lane>=16), N = lane&15)
  const int col = nt * 16 + (lane & 15);
  const int r0  = mt0 * 16 + (lane >> 4) * 8;
#pragma unroll
  for (int i = 0; i < 8; ++i) {
    gates[(size_t)(r0 + i) * G4H + col]      = acc0[i];
    gates[(size_t)(r0 + 16 + i) * G4H + col] = acc1[i];
  }
}

// LayerNorm over 4096 gates of one batch row + LSTM cell update.
// One block (256 threads) per row. Gate order: [i | f | c | o].
__device__ __forceinline__ void ln_cell(const float* __restrict__ gates,
                                        const float* __restrict__ bi,
                                        const float* __restrict__ bh,
                                        const float* __restrict__ gam,
                                        const float* __restrict__ bet,
                                        float* __restrict__ c_st,
                                        float* __restrict__ h_st,
                                        __bf16* __restrict__ hb,
                                        float* __restrict__ outRow, int row) {
  __shared__ float ls[8], lq[8], stats[2];
  float v[16];
  float s = 0.f, q = 0.f;
  const float* gr = gates + (size_t)row * G4H;
#pragma unroll
  for (int j = 0; j < 16; ++j) {
    int   idx = threadIdx.x + j * NTHR;
    float g   = gr[idx] + bi[idx] + bh[idx];
    v[j] = g;
    s += g;
    q += g * g;
  }
#pragma unroll
  for (int off = 16; off > 0; off >>= 1) {
    s += __shfl_down(s, off, 32);
    q += __shfl_down(q, off, 32);
  }
  if ((threadIdx.x & 31) == 0) {
    ls[threadIdx.x >> 5] = s;
    lq[threadIdx.x >> 5] = q;
  }
  __syncthreads();
  if (threadIdx.x == 0) {
    float S = 0.f, Q = 0.f;
#pragma unroll
    for (int i = 0; i < 8; ++i) { S += ls[i]; Q += lq[i]; }
    float mu  = S * (1.0f / G4H);
    float var = Q * (1.0f / G4H) - mu * mu;
    stats[0]  = mu;
    stats[1]  = rsqrtf(var + 1e-5f);
  }
  __syncthreads();
  const float mu = stats[0], rstd = stats[1];
#pragma unroll
  for (int jj = 0; jj < 4; ++jj) {
    int   u  = threadIdx.x + jj * NTHR;
    float ig = (v[jj]      - mu) * rstd * gam[u]        + bet[u];
    float fg = (v[jj + 4]  - mu) * rstd * gam[u + 1024] + bet[u + 1024];
    float cg = (v[jj + 8]  - mu) * rstd * gam[u + 2048] + bet[u + 2048];
    float og = (v[jj + 12] - mu) * rstd * gam[u + 3072] + bet[u + 3072];
    size_t si = (size_t)row * HID + u;
    float  cn = sigmoidf_(fg) * c_st[si] + sigmoidf_(ig) * tanhf(cg);
    float  hn = sigmoidf_(og) * tanhf(cn);
    c_st[si] = cn;
    h_st[si] = hn;
    hb[si]   = (__bf16)hn;
    if (outRow) outRow[u] = hn;
  }
  __syncthreads();
}

__global__ __launch_bounds__(NTHR) void lstm_persistent(
    const __bf16* __restrict__ xb,
    const __bf16* __restrict__ Wi0t, const __bf16* __restrict__ Wh0t,
    const __bf16* __restrict__ Wi1t, const __bf16* __restrict__ Wh1t,
    const float* __restrict__ bi0, const float* __restrict__ bh0,
    const float* __restrict__ g0, const float* __restrict__ be0,
    const float* __restrict__ bi1, const float* __restrict__ bh1,
    const float* __restrict__ g1, const float* __restrict__ be1,
    float* h0, float* c0, float* h1, float* c1,
    __bf16* h0b, __bf16* h1b,
    float* gates0, float* gates1,
    unsigned* bar, float* out) {
  unsigned  epoch = 0;
  const int row   = blockIdx.x;
  for (int t = 0; t < T_STEPS; ++t) {
    // Layer 0: gates0 = x_t @ Wi0 + h0 @ Wh0   (biases folded into LN phase)
    gemm_pair<DIN, HID>(xb + (size_t)t * DIN, T_STEPS * DIN, h0b, HID,
                        Wi0t, Wh0t, gates0);
    grid_barrier(bar, epoch);
    ln_cell(gates0, bi0, bh0, g0, be0, c0, h0, h0b, nullptr, row);
    grid_barrier(bar, epoch);
    // Layer 1: gates1 = h0 @ Wi1 + h1 @ Wh1
    gemm_pair<HID, HID>(h0b, HID, h1b, HID, Wi1t, Wh1t, gates1);
    grid_barrier(bar, epoch);
    ln_cell(gates1, bi1, bh1, g1, be1, c1, h1, h1b,
            out + ((size_t)row * T_STEPS + t) * HID, row);
    grid_barrier(bar, epoch);
  }
  // Final states: outputs | h_stack[2,B,H] | c_stack[2,B,H]
  const size_t HS = (size_t)BATCH * T_STEPS * HID;
  const size_t CS = HS + 2u * (size_t)BATCH * HID;
#pragma unroll
  for (int jj = 0; jj < 4; ++jj) {
    int    u  = threadIdx.x + jj * NTHR;
    size_t si = (size_t)row * HID + u;
    out[HS + si]                           = h0[si];
    out[HS + (size_t)BATCH * HID + si]     = h1[si];
    out[CS + si]                           = c0[si];
    out[CS + (size_t)BATCH * HID + si]     = c1[si];
  }
}

// --- setup kernels (graph-capture safe, deterministic, rerun every call) ----

// w:[K,4096] fp32 row-major  ->  wt:[4096,K] bf16 row-major (LDS tile transpose)
__global__ __launch_bounds__(256) void k_w_cvt(const float* __restrict__ w,
                                               __bf16* __restrict__ wt, int K) {
  __shared__ float tile[32][33];
  const int n0 = blockIdx.x * 32;
  const int k0 = blockIdx.y * 32;
#pragma unroll
  for (int i = threadIdx.y; i < 32; i += 8)
    tile[i][threadIdx.x] = w[(size_t)(k0 + i) * G4H + n0 + threadIdx.x];
  __syncthreads();
#pragma unroll
  for (int i = threadIdx.y; i < 32; i += 8)
    wt[(size_t)(n0 + i) * K + k0 + threadIdx.x] = (__bf16)tile[threadIdx.x][i];
}

__global__ __launch_bounds__(256) void k_x_cvt(const float* __restrict__ x,
                                               __bf16* __restrict__ xb, int n) {
  for (int i = blockIdx.x * blockDim.x + threadIdx.x; i < n;
       i += gridDim.x * blockDim.x)
    xb[i] = (__bf16)x[i];
}

__global__ __launch_bounds__(256) void k_state_init(unsigned* bar, float* h0,
                                                    float* c0, float* h1,
                                                    float* c1, __bf16* h0b,
                                                    __bf16* h1b) {
  int i = blockIdx.x * blockDim.x + threadIdx.x;  // 65536 = BATCH*HID
  if (i == 0) bar[0] = 0u;
  if (i < BATCH * HID) {
    h0[i]  = 0.f;
    c0[i]  = 0.f;
    h1[i]  = 0.f;
    c1[i]  = 0.f;
    h0b[i] = (__bf16)0.f;
    h1b[i] = (__bf16)0.f;
  }
}

// ---------------------------------------------------------------------------

extern "C" void kernel_launch(void* const* d_in, const int* in_sizes, int n_in,
                              void* d_out, int out_size, void* d_ws,
                              size_t ws_size, hipStream_t stream) {
  (void)in_sizes; (void)n_in; (void)out_size; (void)ws_size;
  const float* x   = (const float*)d_in[0];
  const float* Wi0 = (const float*)d_in[1];
  const float* bi0 = (const float*)d_in[2];
  const float* Wh0 = (const float*)d_in[3];
  const float* bh0 = (const float*)d_in[4];
  const float* g0  = (const float*)d_in[5];
  const float* be0 = (const float*)d_in[6];
  const float* Wi1 = (const float*)d_in[7];
  const float* bi1 = (const float*)d_in[8];
  const float* Wh1 = (const float*)d_in[9];
  const float* bh1 = (const float*)d_in[10];
  const float* g1  = (const float*)d_in[11];
  const float* be1 = (const float*)d_in[12];
  float*       out = (float*)d_out;

  char*  ws  = (char*)d_ws;
  size_t off = 0;
  auto take = [&](size_t bytes) -> void* {
    void* p = ws + off;
    off += (bytes + 255) & ~(size_t)255;
    return p;
  };
  unsigned* bar  = (unsigned*)take(256);
  __bf16*   Wi0t = (__bf16*)take((size_t)G4H * DIN * 2);
  __bf16*   Wh0t = (__bf16*)take((size_t)G4H * HID * 2);
  __bf16*   Wi1t = (__bf16*)take((size_t)G4H * HID * 2);
  __bf16*   Wh1t = (__bf16*)take((size_t)G4H * HID * 2);
  __bf16*   xb   = (__bf16*)take((size_t)BATCH * T_STEPS * DIN * 2);
  float*    h0   = (float*)take((size_t)BATCH * HID * 4);
  float*    c0   = (float*)take((size_t)BATCH * HID * 4);
  float*    h1   = (float*)take((size_t)BATCH * HID * 4);
  float*    c1   = (float*)take((size_t)BATCH * HID * 4);
  __bf16*   h0b  = (__bf16*)take((size_t)BATCH * HID * 2);
  __bf16*   h1b  = (__bf16*)take((size_t)BATCH * HID * 2);
  float*    gts0 = (float*)take((size_t)BATCH * G4H * 4);
  float*    gts1 = (float*)take((size_t)BATCH * G4H * 4);

  dim3 tb(32, 8);
  k_w_cvt<<<dim3(G4H / 32, DIN / 32), tb, 0, stream>>>(Wi0, Wi0t, DIN);
  k_w_cvt<<<dim3(G4H / 32, HID / 32), tb, 0, stream>>>(Wh0, Wh0t, HID);
  k_w_cvt<<<dim3(G4H / 32, HID / 32), tb, 0, stream>>>(Wi1, Wi1t, HID);
  k_w_cvt<<<dim3(G4H / 32, HID / 32), tb, 0, stream>>>(Wh1, Wh1t, HID);
  k_x_cvt<<<4096, 256, 0, stream>>>(x, xb, BATCH * T_STEPS * DIN);
  k_state_init<<<256, 256, 0, stream>>>(bar, h0, c0, h1, c1, h0b, h1b);

  lstm_persistent<<<NBLK, NTHR, 0, stream>>>(
      xb, Wi0t, Wh0t, Wi1t, Wh1t, bi0, bh0, g0, be0, bi1, bh1, g1, be1, h0, c0,
      h1, c1, h0b, h1b, gts0, gts1, bar, out);
}
